// TFDebertaV2Attention_42030549959030
// MI455X (gfx1250) — matrix operand order repaired and checked by reading, not compiled, not run on hardware
//
#include <hip/hip_runtime.h>
#include <hip/hip_fp16.h>

// ---------------- problem constants (match reference) ----------------
constexpr int kB    = 2;
constexpr int kL    = 1024;
constexpr int kHID  = 1024;
constexpr int kNH   = 16;
constexpr int kHEAD = 64;
constexpr int kS2   = 512;            // 2 * ATT_SPAN
constexpr int kM    = kB * kL;        // 2048 rows of hidden_states

typedef __attribute__((ext_vector_type(16))) _Float16 v16h;
typedef __attribute__((ext_vector_type(8)))  float    v8f;
typedef int v4i_ld __attribute__((vector_size(4 * sizeof(int))));

#if defined(__AMDGCN__) && __has_builtin(__builtin_amdgcn_global_load_async_to_lds_b128)
#define HAS_ASYNC_LDS 1
#else
#define HAS_ASYNC_LDS 0
#endif

static __device__ __forceinline__ int lane_id(){ return (int)(threadIdx.x & 31u); }
static __device__ __forceinline__ int wave_id(){ return (int)(threadIdx.x >> 5u); }

static __device__ __forceinline__ void wait_async0(){
#if defined(__AMDGCN__)
#if __has_builtin(__builtin_amdgcn_s_wait_asynccnt)
  __builtin_amdgcn_s_wait_asynccnt(0);
#else
  asm volatile("s_wait_asynccnt 0x0" ::: "memory");
#endif
#endif
}

static __device__ __forceinline__ v8f wmma_f16(v16h a, v16h b, v8f c){
  // D = A(16x32 f16) x B(32x16 f16) + C(16x16 f32)
  return __builtin_amdgcn_wmma_f32_16x16x32_f16(false, a, false, b, (short)0, c, false, false);
}

// ---- WMMA fragment loaders (wave32), contiguous k order k = 16*(l/16) + s in both A and B ----
// A 16x32 f16: lane l -> row m=l&15; kb = (l<16)?0:16; holds K={kb..kb+15}
static __device__ __forceinline__ v16h frag_a_f16(const _Float16* base, int ld){
  int l = lane_id(); int m = l & 15; int kb = (l < 16) ? 0 : 16;
  const _Float16* p = base + (size_t)m * ld + kb;
  v16h a;
  #pragma unroll
  for (int i = 0; i < 8; ++i){ a[i] = p[i]; a[i + 8] = p[i + 8]; }
  return a;
}
// Same A fragment but source is f32 (global or LDS), converted to f16
static __device__ __forceinline__ v16h frag_a_f32(const float* base, int ld){
  int l = lane_id(); int m = l & 15; int kb = (l < 16) ? 0 : 16;
  const float* p = base + (size_t)m * ld + kb;
  v16h a;
  #pragma unroll
  for (int i = 0; i < 8; ++i){ a[i] = (_Float16)p[i]; a[i + 8] = (_Float16)p[i + 8]; }
  return a;
}
// B 32x16 f16 where B(k,n) = src[n*ld + k]  (column n of B contiguous in src row n)
static __device__ __forceinline__ v16h frag_b_T(const _Float16* base, int ld){
  int l = lane_id(); int n = l & 15; int kb = (l < 16) ? 0 : 16;
  const _Float16* p = base + (size_t)n * ld + kb;
  v16h b;
  #pragma unroll
  for (int i = 0; i < 16; ++i) b[i] = p[i];
  return b;
}
// B 32x16 from f32 LDS tile stored [k][kWS] (row-major, no transpose), converted to f16
constexpr int kWS = 68;   // f32 row stride of staged W chunk (16B-aligned rows)
static __device__ __forceinline__ v16h frag_b_kxn_f32(const float* buf, int n_off){
  int l = lane_id(); int n = (l & 15) + n_off; int kb = (l < 16) ? 0 : 16;
  v16h b;
  #pragma unroll
  for (int i = 0; i < 16; ++i) b[i] = (_Float16)buf[(kb + i) * kWS + n];
  return b;
}
// C/D 16x16 f32: lane l -> col=l&15, element i -> row i + 8*(l>=16)

// ---- async (ASYNCcnt-tracked) staging of a 32x64 f32 W chunk into LDS [k][kWS] ----
static __device__ __forceinline__ void stage_w(const float* __restrict__ Wg, float* buf){
  int t = (int)threadIdx.x;
  #pragma unroll
  for (int j = 0; j < 2; ++j){
    int idx = t + 256 * j;        // b128 chunk id, 0..511 (2048 f32 total)
    int k = idx >> 4;             // row 0..31
    int c = (idx & 15) << 2;      // col 0,4,..,60
    const float* src = Wg + (size_t)k * kHID + c;
    float* dst = buf + k * kWS + c;
#if HAS_ASYNC_LDS
    __builtin_amdgcn_global_load_async_to_lds_b128(
        (__attribute__((address_space(1))) v4i_ld*)src,
        (__attribute__((address_space(3))) v4i_ld*)dst, 0, 0);
#else
    *(float4*)dst = *(const float4*)src;
#endif
  }
}

// ---------------- bucket index table (replicates reference log-bucket) ----------------
__global__ void bucket_init(int* __restrict__ tbl){
  int i = (int)(blockIdx.x * blockDim.x + threadIdx.x);
  if (i >= 2047) return;
  int rel = i - 1023;                       // rel = q - k
  const int mid = 128;                      // BUCKETS/2
  int sgn = (rel > 0) ? 1 : ((rel < 0) ? -1 : 0);
  bool in_small = (rel < mid) && (rel > -mid);
  int abs_pos = in_small ? (mid - 1) : (rel < 0 ? -rel : rel);
  float log_pos = ceilf(logf((float)abs_pos / 128.0f) /
                        logf(511.0f / 128.0f) * 127.0f) + 128.0f;
  int bucket = (abs_pos <= mid) ? rel : (int)(log_pos * (float)sgn);
  int s = bucket + 256;                     // + ATT_SPAN
  s = (s < 0) ? 0 : ((s > 511) ? 511 : s);
  tbl[i] = s;
}

// ---------------- GEMM: C[M,1024] = A_f32[M,1024] @ W_f32[1024,1024] + bias ----------------
// Double-buffered async W staging; writes C (f16 row-major) and/or Ct (f16 [N][M]).
__global__ __launch_bounds__(256) void gemm_xw_f32(
    const float* __restrict__ A, const float* __restrict__ W, const float* __restrict__ bias,
    _Float16* __restrict__ C, _Float16* __restrict__ Ct, int M){
  __shared__ float wbuf[2][32 * kWS];       // 2 x 8.5 KB
  int n0 = blockIdx.x * 64, m0 = blockIdx.y * 64;
  int w = wave_id(), l = lane_id();
  int tm = w >> 1, tn0 = (w & 1) * 2;       // wave -> (row tile, 2 col tiles)
  const float* Arow = A + (size_t)(m0 + tm * 16) * kHID;
  v8f acc0 = {}, acc1 = {};
  stage_w(W + n0, &wbuf[0][0]);
  wait_async0();
  __syncthreads();
  for (int kk = 0; kk < kHID; kk += 32){
    int cur = (kk >> 5) & 1;
    if (kk + 32 < kHID){
      stage_w(W + (size_t)(kk + 32) * kHID + n0, &wbuf[cur ^ 1][0]);
      __builtin_prefetch(Arow + (size_t)(l & 15) * kHID + kk + 32, 0, 0);
    }
    v16h a = frag_a_f32(Arow + kk, kHID);
    acc0 = wmma_f16(a, frag_b_kxn_f32(&wbuf[cur][0], tn0 * 16), acc0);
    acc1 = wmma_f16(a, frag_b_kxn_f32(&wbuf[cur][0], tn0 * 16 + 16), acc1);
    wait_async0();
    __syncthreads();
  }
  int col = l & 15, rbase = (l >> 4) * 8;
  #pragma unroll
  for (int i = 0; i < 8; ++i){
    int m = m0 + tm * 16 + rbase + i;
    int n = n0 + tn0 * 16 + col;
    float v0 = acc0[i] + bias[n];
    float v1 = acc1[i] + bias[n + 16];
    if (C){ C[(size_t)m * kHID + n] = (_Float16)v0; C[(size_t)m * kHID + n + 16] = (_Float16)v1; }
    if (Ct){ Ct[(size_t)n * M + m] = (_Float16)v0; Ct[(size_t)(n + 16) * M + m] = (_Float16)v1; }
  }
}

// ---------------- bias tables: T[b,h,q,s] = sum_d Qlike[b,q,h*64+d] * Plike[s,h*64+d] ----------------
__global__ __launch_bounds__(256) void gemm_bias_table(
    const _Float16* __restrict__ Qlike,   // [B*L, HID] f16
    const _Float16* __restrict__ Plike,   // [512, HID] f16
    _Float16* __restrict__ T){            // [B,NH,L,512] f16
  int s0 = blockIdx.x * 64, q0 = blockIdx.y * 64;
  int bh = blockIdx.z; int b = bh / kNH, h = bh % kNH;
  int w = wave_id(), l = lane_id();
  int tm = w >> 1, tn0 = (w & 1) * 2;
  v8f acc0 = {}, acc1 = {};
  #pragma unroll
  for (int ks = 0; ks < 2; ++ks){           // K = HEAD = 64 -> 2 chunks of 32
    v16h a  = frag_a_f16(Qlike + (size_t)(b * kL + q0 + tm * 16) * kHID + h * kHEAD + ks * 32, kHID);
    v16h b0 = frag_b_T(Plike + (size_t)(s0 + tn0 * 16) * kHID + h * kHEAD + ks * 32, kHID);
    v16h b1 = frag_b_T(Plike + (size_t)(s0 + tn0 * 16 + 16) * kHID + h * kHEAD + ks * 32, kHID);
    acc0 = wmma_f16(a, b0, acc0);
    acc1 = wmma_f16(a, b1, acc1);
  }
  int col = l & 15, rbase = (l >> 4) * 8;
  size_t base = (size_t)bh * kL * kS2;
  #pragma unroll
  for (int i = 0; i < 8; ++i){
    int q = q0 + tm * 16 + rbase + i;
    T[base + (size_t)q * kS2 + s0 + tn0 * 16 + col]       = (_Float16)acc0[i];
    T[base + (size_t)q * kS2 + s0 + (tn0 + 1) * 16 + col] = (_Float16)acc1[i];
  }
}

// ---------------- fused attention: scores + bias gather + mask + softmax + P@V ----------------
constexpr int kLDP = 1036;  // padded f32 row stride (1036 % 64 == 12, decorrelates banks)
__global__ __launch_bounds__(256) void attn_kernel(
    const _Float16* __restrict__ Qh, const _Float16* __restrict__ Kh,
    const _Float16* __restrict__ Vt,          // [HID][B*L] f16 (V transposed)
    const _Float16* __restrict__ C2P, const _Float16* __restrict__ P2C, // [B,NH,L,512]
    const int* __restrict__ mask, const int* __restrict__ tbl,
    _Float16* __restrict__ CTX){
  extern __shared__ float S[];                // [64][kLDP] scores/probs
  int gz = (int)blockIdx.x;
  int qb = gz & 15; int bh = gz >> 4; int b = bh / kNH, h = bh % kNH;
  int q0 = qb * 64;
  int w = wave_id(), l = lane_id();
  int col = l & 15, rbase = (l >> 4) * 8;
  const float inv_scale = 0.07216878365f;     // 1/sqrt(HEAD * 3)

  // ---- phase 1: S[q,k] = (q.k + c2p + p2c) * inv_scale, masked ----
  int tm = w >> 1;
  const _Float16* Abase = Qh + (size_t)(b * kL + q0 + tm * 16) * kHID + h * kHEAD;
  v16h a0 = frag_a_f16(Abase, kHID);
  v16h a1 = frag_a_f16(Abase + 32, kHID);
  size_t biasB = (size_t)bh * kL * kS2;
  for (int jt = 0; jt < 32; ++jt){
    int tn = (w & 1) * 32 + jt;               // 64 column tiles split over wave pairs
    int k0 = tn * 16;
    v8f acc = {};
    acc = wmma_f16(a0, frag_b_T(Kh + (size_t)(b * kL + k0) * kHID + h * kHEAD,      kHID), acc);
    acc = wmma_f16(a1, frag_b_T(Kh + (size_t)(b * kL + k0) * kHID + h * kHEAD + 32, kHID), acc);
    #pragma unroll
    for (int i = 0; i < 8; ++i){
      int qr = tm * 16 + rbase + i;           // row within q-block
      int q  = q0 + qr;
      int k  = k0 + col;
      int s1 = tbl[(q - k) + 1023];           // serves both c2p and p2c (bucket is odd)
      float c2p = (float)C2P[biasB + (size_t)q * kS2 + s1];
      float p2c = (float)P2C[biasB + (size_t)k * kS2 + s1];
      int mk = mask[((size_t)b * kL + q) * kL + k];
      float v = (acc[i] + c2p + p2c) * inv_scale;
      S[qr * kLDP + k] = mk ? v : -1e30f;
    }
  }
  __syncthreads();

  // ---- phase 2: row softmax (wave w owns rows w*8 .. w*8+7) ----
  for (int r = w * 8; r < w * 8 + 8; ++r){
    float mx = -1e30f;
    for (int j = l; j < kL; j += 32) mx = fmaxf(mx, S[r * kLDP + j]);
    #pragma unroll
    for (int off = 16; off; off >>= 1) mx = fmaxf(mx, __shfl_xor(mx, off, 32));
    float sum = 0.f;
    for (int j = l; j < kL; j += 32){
      float e = expf(S[r * kLDP + j] - mx);
      S[r * kLDP + j] = e;
      sum += e;
    }
    #pragma unroll
    for (int off = 16; off; off >>= 1) sum += __shfl_xor(sum, off, 32);
    float rs = (sum > 0.f) ? (1.f / sum) : 0.f;
    for (int j = l; j < kL; j += 32) S[r * kLDP + j] *= rs;
  }
  __syncthreads();

  // ---- phase 3: ctx[64,64] = P[64,1024] @ V[1024,64] ----
  int tnc0 = (w & 1) * 2;
  v8f c0 = {}, c1 = {};
  for (int kk = 0; kk < kL; kk += 32){
    v16h a = frag_a_f32(S + (size_t)(tm * 16) * kLDP + kk, kLDP);
    v16h b0 = frag_b_T(Vt + (size_t)(h * kHEAD + tnc0 * 16)      * kM + (size_t)b * kL + kk, kM);
    v16h b1 = frag_b_T(Vt + (size_t)(h * kHEAD + tnc0 * 16 + 16) * kM + (size_t)b * kL + kk, kM);
    c0 = wmma_f16(a, b0, c0);
    c1 = wmma_f16(a, b1, c1);
  }
  #pragma unroll
  for (int i = 0; i < 8; ++i){
    int q = q0 + tm * 16 + rbase + i;
    size_t o = ((size_t)b * kL + q) * kHID + h * kHEAD;
    CTX[o + tnc0 * 16 + col]        = (_Float16)c0[i];
    CTX[o + (tnc0 + 1) * 16 + col]  = (_Float16)c1[i];
  }
}

// ---------------- output projection + residual: Y = CTX@Wo + bo + X (f32) ----------------
__global__ __launch_bounds__(256) void gemm_out(
    const _Float16* __restrict__ CTX, const float* __restrict__ Wo,
    const float* __restrict__ bo, const float* __restrict__ X, float* __restrict__ Y){
  __shared__ float wbuf[2][32 * kWS];
  int n0 = blockIdx.x * 64, m0 = blockIdx.y * 64;
  int w = wave_id(), l = lane_id();
  int tm = w >> 1, tn0 = (w & 1) * 2;
  const _Float16* Arow = CTX + (size_t)(m0 + tm * 16) * kHID;
  v8f acc0 = {}, acc1 = {};
  stage_w(Wo + n0, &wbuf[0][0]);
  wait_async0();
  __syncthreads();
  for (int kk = 0; kk < kHID; kk += 32){
    int cur = (kk >> 5) & 1;
    if (kk + 32 < kHID){
      stage_w(Wo + (size_t)(kk + 32) * kHID + n0, &wbuf[cur ^ 1][0]);
      __builtin_prefetch(Arow + (size_t)(l & 15) * kHID + kk + 32, 0, 0);
    }
    v16h a = frag_a_f16(Arow + kk, kHID);
    acc0 = wmma_f16(a, frag_b_kxn_f32(&wbuf[cur][0], tn0 * 16), acc0);
    acc1 = wmma_f16(a, frag_b_kxn_f32(&wbuf[cur][0], tn0 * 16 + 16), acc1);
    wait_async0();
    __syncthreads();
  }
  int col = l & 15, rbase = (l >> 4) * 8;
  #pragma unroll
  for (int i = 0; i < 8; ++i){
    int m = m0 + tm * 16 + rbase + i;
    int n = n0 + tn0 * 16 + col;
    Y[(size_t)m * kHID + n]      = acc0[i] + bo[n]      + X[(size_t)m * kHID + n];
    Y[(size_t)m * kHID + n + 16] = acc1[i] + bo[n + 16] + X[(size_t)m * kHID + n + 16];
  }
}

// ---------------- row LayerNorm -> d_out (f32) ----------------
__global__ __launch_bounds__(256) void ln_kernel(
    const float* __restrict__ Y, const float* __restrict__ gamma,
    const float* __restrict__ beta, float* __restrict__ out){
  __shared__ float rs[8], rs2[8];
  int row = (int)blockIdx.x;
  const float* y = Y + (size_t)row * kHID;
  float v[4], s = 0.f, s2 = 0.f;
  #pragma unroll
  for (int j = 0; j < 4; ++j){
    v[j] = y[threadIdx.x + 256 * j];
    s += v[j]; s2 += v[j] * v[j];
  }
  #pragma unroll
  for (int off = 16; off; off >>= 1){ s += __shfl_xor(s, off, 32); s2 += __shfl_xor(s2, off, 32); }
  if (lane_id() == 0){ rs[wave_id()] = s; rs2[wave_id()] = s2; }
  __syncthreads();
  float ts = 0.f, ts2 = 0.f;
  #pragma unroll
  for (int i = 0; i < 8; ++i){ ts += rs[i]; ts2 += rs2[i]; }
  float mu = ts * (1.0f / kHID);
  float var = ts2 * (1.0f / kHID) - mu * mu;
  float rstd = rsqrtf(var + 1e-7f);
  #pragma unroll
  for (int j = 0; j < 4; ++j){
    int c = (int)threadIdx.x + 256 * j;
    out[(size_t)row * kHID + c] = (v[j] - mu) * rstd * gamma[c] + beta[c];
  }
}

// ---------------- host launcher ----------------
extern "C" void kernel_launch(void* const* d_in, const int* in_sizes, int n_in,
                              void* d_out, int out_size, void* d_ws, size_t ws_size,
                              hipStream_t stream) {
  (void)in_sizes; (void)n_in; (void)out_size; (void)ws_size;
  const float* X      = (const float*)d_in[0];   // hidden_states [B,L,HID]
  const int*   mask   = (const int*)  d_in[1];   // attention_mask [B,1,L,L]
  const float* relemb = (const float*)d_in[2];   // rel_embeddings [512,HID]
  const float* Wq = (const float*)d_in[3];  const float* bq = (const float*)d_in[4];
  const float* Wk = (const float*)d_in[5];  const float* bk = (const float*)d_in[6];
  const float* Wv = (const float*)d_in[7];  const float* bv = (const float*)d_in[8];
  const float* Wo = (const float*)d_in[9];  const float* bo = (const float*)d_in[10];
  const float* ga = (const float*)d_in[11]; const float* be = (const float*)d_in[12];

  constexpr size_t MB = 1024u * 1024u;
  char* ws = (char*)d_ws;
  _Float16* Qh  = (_Float16*)(ws + 0 * MB);    //  4 MB  Q f16 [B*L,HID]
  _Float16* Kh  = (_Float16*)(ws + 4 * MB);    //  4 MB  K f16
  _Float16* Vt  = (_Float16*)(ws + 8 * MB);    //  4 MB  V^T f16 [HID][B*L]
  _Float16* PK  = (_Float16*)(ws + 12 * MB);   //  1 MB  pos_key f16 [512,HID]
  _Float16* PQ  = (_Float16*)(ws + 13 * MB);   //  1 MB  pos_query f16
  _Float16* C2P = (_Float16*)(ws + 14 * MB);   // 32 MB  [B,NH,L,512]
  _Float16* P2C = (_Float16*)(ws + 46 * MB);   // 32 MB  [B,NH,L,512]
  _Float16* CTX = (_Float16*)(ws + 78 * MB);   //  4 MB  context f16 [B*L,HID]
  float*    Yb  = (float*)   (ws + 82 * MB);   //  8 MB  pre-LN f32
  int*      tbl = (int*)     (ws + 90 * MB);   //  8 KB  rel->bucket index

  dim3 blk(256);
  bucket_init<<<dim3(8), blk, 0, stream>>>(tbl);

  // projections (f32 in, f16 out); V written transposed only
  gemm_xw_f32<<<dim3(16, 32), blk, 0, stream>>>(X,      Wq, bq, Qh,      nullptr, kM);
  gemm_xw_f32<<<dim3(16, 32), blk, 0, stream>>>(X,      Wk, bk, Kh,      nullptr, kM);
  gemm_xw_f32<<<dim3(16, 32), blk, 0, stream>>>(X,      Wv, bv, nullptr, Vt,      kM);
  gemm_xw_f32<<<dim3(16, 8),  blk, 0, stream>>>(relemb, Wk, bk, PK,      nullptr, kS2);
  gemm_xw_f32<<<dim3(16, 8),  blk, 0, stream>>>(relemb, Wq, bq, PQ,      nullptr, kS2);

  // disentangled bias tables (L2 resident)
  gemm_bias_table<<<dim3(8, 16, 32), blk, 0, stream>>>(Qh, PK, C2P);
  gemm_bias_table<<<dim3(8, 16, 32), blk, 0, stream>>>(Kh, PQ, P2C);

  // fused attention: 64 q-rows per WG, 259 KB dynamic LDS (CDNA5: 320 KB/WGP)
  size_t shmem = (size_t)64 * kLDP * sizeof(float);
  (void)hipFuncSetAttribute((const void*)attn_kernel,
                            hipFuncAttributeMaxDynamicSharedMemorySize, (int)shmem);
  attn_kernel<<<dim3(kB * kNH * (kL / 64)), blk, shmem, stream>>>(
      Qh, Kh, Vt, C2P, P2C, mask, tbl, CTX);

  // output projection + residual, then LayerNorm
  gemm_out<<<dim3(16, 32), blk, 0, stream>>>(CTX, Wo, bo, X, Yb);
  ln_kernel<<<dim3(kM), blk, 0, stream>>>(Yb, ga, be, (float*)d_out);
}
